// MultiHeadAttention_24696061952258
// MI455X (gfx1250) — matrix operand order repaired and checked by reading, not compiled, not run on hardware
//
#include <hip/hip_runtime.h>
#include <math.h>

// ---------------------------------------------------------------------------
// MultiHeadAttention for MI455X (gfx1250): bf16 WMMA pipeline, fp32 accumulate
// ---------------------------------------------------------------------------

typedef __bf16 bf16_t;
typedef __attribute__((ext_vector_type(16))) __bf16 v16bf;
typedef __attribute__((ext_vector_type(8)))  __bf16 v8bf;
typedef __attribute__((ext_vector_type(8)))  float  v8f;
typedef __attribute__((ext_vector_type(4)))  int    v4i;

#define S_LEN  2048
#define BATCH  2
#define DMODEL 1024
#define NHEADS 16
#define DK     64
#define MROWS  (S_LEN * BATCH)   // 4096 flattened (s,b) rows

#define AS_GLOBAL __attribute__((address_space(1)))
#define AS_LOCAL  __attribute__((address_space(3)))

#if __has_builtin(__builtin_amdgcn_global_load_async_to_lds_b128) && \
    __has_builtin(__builtin_amdgcn_s_wait_asynccnt)
#define HAVE_ASYNC_LDS 1
#else
#define HAVE_ASYNC_LDS 0
#endif

// ---- 16B global -> LDS copy: async (ASYNCcnt) when available --------------
__device__ __forceinline__ void cp128_g2l(const bf16_t* g, bf16_t* l) {
#if HAVE_ASYNC_LDS
  __builtin_amdgcn_global_load_async_to_lds_b128(
      (AS_GLOBAL v4i*)g, (AS_LOCAL v4i*)l, 0, 0);
#else
  *(uint4*)l = *(const uint4*)g;
#endif
}

__device__ __forceinline__ void async_wait_all() {
#if HAVE_ASYNC_LDS
  __builtin_amdgcn_s_wait_asynccnt(0);
#endif
}

// ---- WMMA wrapper: D = A(16x32 bf16) * B(32x16 bf16) + C(16x16 f32) --------
__device__ __forceinline__ v8f wmma_bf16f32(v16bf a, v16bf b, v8f c) {
  return __builtin_amdgcn_wmma_f32_16x16x32_bf16(
      /*neg_a=*/false, a, /*neg_b=*/false, b,
      /*c_mod=*/(short)0, c, /*reuse_a=*/false, /*reuse_b=*/false);
}

// ---- A fragment (16x32 bf16) from LDS, row-major rows m, stride ld ---------
// ISA 7.12.2: lane m = lane%16; lanes 0-15 hold K 0..7 & 16..23,
//             lanes 16-31 hold K 8..15 & 24..31 (two 16B contiguous reads).
__device__ __forceinline__ v16bf frag_a(const bf16_t* base, int ld, int lane) {
  const int m  = lane & 15;
  const int kb = (lane < 16) ? 0 : 8;
  const bf16_t* p = base + m * ld + kb;
  v8bf lo = *(const v8bf*)(p);
  v8bf hi = *(const v8bf*)(p + 16);
  v16bf a;
#pragma unroll
  for (int i = 0; i < 8; ++i) { a[i] = lo[i]; a[i + 8] = hi[i]; }
  return a;
}

// ---- B fragment (32x16 bf16) from LDS stored as [n][k] (k contiguous) ------
// lane n = lane%16; lanes 0-15 hold K 0..7 & 16..23, lanes 16-31 hold K 8..15 & 24..31.
__device__ __forceinline__ v16bf frag_b(const bf16_t* base, int ld, int lane) {
  const int n  = lane & 15;
  const int kb = (lane < 16) ? 0 : 8;
  const bf16_t* p = base + n * ld + kb;
  v8bf lo = *(const v8bf*)(p);
  v8bf hi = *(const v8bf*)(p + 16);
  v16bf b;
#pragma unroll
  for (int i = 0; i < 8; ++i) { b[i] = lo[i]; b[i + 8] = hi[i]; }
  return b;
}

// convert 16 contiguous fp32 -> 16 bf16 into LDS
__device__ __forceinline__ void cvt16_to_lds(bf16_t* dst, const float4* gp) {
  float4 f0 = gp[0], f1 = gp[1], f2 = gp[2], f3 = gp[3];
  v8bf p0, p1;
  p0[0]=(bf16_t)f0.x; p0[1]=(bf16_t)f0.y; p0[2]=(bf16_t)f0.z; p0[3]=(bf16_t)f0.w;
  p0[4]=(bf16_t)f1.x; p0[5]=(bf16_t)f1.y; p0[6]=(bf16_t)f1.z; p0[7]=(bf16_t)f1.w;
  p1[0]=(bf16_t)f2.x; p1[1]=(bf16_t)f2.y; p1[2]=(bf16_t)f2.z; p1[3]=(bf16_t)f2.w;
  p1[4]=(bf16_t)f3.x; p1[5]=(bf16_t)f3.y; p1[6]=(bf16_t)f3.z; p1[7]=(bf16_t)f3.w;
  *(v8bf*)(dst)     = p0;
  *(v8bf*)(dst + 8) = p1;
}

// ---------------------------------------------------------------------------
// GEMM: Out[M,N] = A[M,K] @ W[N,K]^T + bias   (nn.Linear convention)
// 128x64 tile / block (4 waves, each 32x64 = 8 WMMA per 32-K step).
// ---------------------------------------------------------------------------
template <bool IN_BF16, bool OUT_F32>
__global__ __launch_bounds__(128)
void gemm_xwt(const void* __restrict__ Ain,
              const float* __restrict__ W,
              const float* __restrict__ bias,
              void* __restrict__ Out,
              int K, int N)
{
  __shared__ bf16_t As[128][32];  // [m][k]
  __shared__ bf16_t Bs[64][32];   // [n][k]  (W rows are k-contiguous already)

  const int tid  = threadIdx.x;
  const int lane = tid & 31;
  const int wave = tid >> 5;
  const int m0 = blockIdx.x * 128;
  const int n0 = blockIdx.y * 64;

  v8f acc[2][4];
  {
    v8f z = {};
#pragma unroll
    for (int mt = 0; mt < 2; ++mt)
#pragma unroll
      for (int nt = 0; nt < 4; ++nt) acc[mt][nt] = z;
  }

  const int wrow = tid >> 1;         // 0..63 (W tile row)
  const int wseg = (tid & 1) * 16;   // 0 or 16

  for (int k0 = 0; k0 < K; k0 += 32) {
    // ---- stage A tile: one 32-elem row per thread ----
    if (IN_BF16) {
      const bf16_t* A = (const bf16_t*)Ain;
      const bf16_t* src = A + (size_t)(m0 + tid) * K + k0;
      cp128_g2l(src,      &As[tid][0]);
      cp128_g2l(src + 8,  &As[tid][8]);
      cp128_g2l(src + 16, &As[tid][16]);
      cp128_g2l(src + 24, &As[tid][24]);
      if (k0 + 32 < K) __builtin_prefetch(src + 32, 0, 1);
    } else {
      const float* A = (const float*)Ain;
      const float* src = A + (size_t)(m0 + tid) * K + k0;
      cvt16_to_lds(&As[tid][0],  (const float4*)(src));
      cvt16_to_lds(&As[tid][16], (const float4*)(src + 16));
      if (k0 + 32 < K) __builtin_prefetch(src + 32, 0, 1);
    }
    // ---- stage W tile (fp32 -> bf16) ----
    {
      const float* src = W + (size_t)(n0 + wrow) * K + k0 + wseg;
      cvt16_to_lds(&Bs[wrow][wseg], (const float4*)src);
      if (k0 + 32 < K) __builtin_prefetch(src + 32, 0, 1);
    }
    async_wait_all();
    __syncthreads();

    v16bf a[2];
#pragma unroll
    for (int mt = 0; mt < 2; ++mt)
      a[mt] = frag_a(&As[wave * 32 + mt * 16][0], 32, lane);
#pragma unroll
    for (int nt = 0; nt < 4; ++nt) {
      const v16bf b = frag_b(&Bs[nt * 16][0], 32, lane);
#pragma unroll
      for (int mt = 0; mt < 2; ++mt)
        acc[mt][nt] = wmma_bf16f32(a[mt], b, acc[mt][nt]);
    }
    __syncthreads();
  }

  // ---- epilogue: C layout (m = r + 8*(lane>=16), n = lane%16) ----
  const int nlo = n0 + (lane & 15);
#pragma unroll
  for (int mt = 0; mt < 2; ++mt) {
    const int mbase = m0 + wave * 32 + mt * 16 + ((lane & 16) ? 8 : 0);
#pragma unroll
    for (int nt = 0; nt < 4; ++nt) {
      const int n = nlo + nt * 16;
      const float bn = bias[n];
#pragma unroll
      for (int r = 0; r < 8; ++r) {
        const float val = acc[mt][nt][r] + bn;
        if (OUT_F32) ((float*)Out)[(size_t)(mbase + r) * N + n] = val;
        else ((bf16_t*)Out)[(size_t)(mbase + r) * N + n] = (bf16_t)val;
      }
    }
  }
}

// ---------------------------------------------------------------------------
// Flash attention per (b,h): Sᵀ = K·Qᵀ (softmax axis lands in VGPR/M dim),
// running-max softmax, Oᵀ += Vᵀ·Pᵀ.  64 query rows / block, 64-key blocks.
// ---------------------------------------------------------------------------
__global__ __launch_bounds__(128)
void flash_attn(const bf16_t* __restrict__ Qg,
                const bf16_t* __restrict__ Kg,
                const bf16_t* __restrict__ Vg,
                bf16_t* __restrict__ Og)
{
  __shared__ bf16_t Qs[64][64];        // [i][d]
  __shared__ bf16_t Ks[64][64];        // [j][d]
  __shared__ bf16_t Vt[64][64];        // [d][j]  (transposed for A-frags)
  __shared__ bf16_t Ps[4][16][64];     // per-wave probs [i][j]

  const int tid  = threadIdx.x;
  const int lane = tid & 31;
  const int wave = tid >> 5;
  const int bh   = blockIdx.y;
  const int b    = bh & (BATCH - 1);
  const int h    = bh >> 1;
  const int i0   = blockIdx.x * 64;
  const int col0 = h * DK;

  // stage Q tile: 64 x 64 bf16 (2 threads/row, 32 elems each, async)
  {
    const int r = tid >> 1;
    const int c = (tid & 1) * 32;
    const bf16_t* src = Qg + ((size_t)(i0 + r) * BATCH + b) * DMODEL + col0 + c;
    cp128_g2l(src,      &Qs[r][c]);
    cp128_g2l(src + 8,  &Qs[r][c + 8]);
    cp128_g2l(src + 16, &Qs[r][c + 16]);
    cp128_g2l(src + 24, &Qs[r][c + 24]);
  }

  float m_run = -1e30f, l_run = 0.0f;
  v8f oacc[4];
  {
    v8f z = {};
#pragma unroll
    for (int i = 0; i < 4; ++i) oacc[i] = z;
  }

  for (int j0 = 0; j0 < S_LEN; j0 += 64) {
    __syncthreads();   // protect Ks/Vt reuse (and Q on first iteration)

    // stage K block: 64 x 64 (2 threads/row, 32 elems, async)
    {
      const int r = tid >> 1;
      const int c = (tid & 1) * 32;
      const bf16_t* src = Kg + ((size_t)(j0 + r) * BATCH + b) * DMODEL + col0 + c;
      cp128_g2l(src,      &Ks[r][c]);
      cp128_g2l(src + 8,  &Ks[r][c + 8]);
      cp128_g2l(src + 16, &Ks[r][c + 16]);
      cp128_g2l(src + 24, &Ks[r][c + 24]);
    }
    // stage V block transposed: Vt[d][j] (needs data reshuffle -> sync path)
    {
      const int jl  = tid & 63;
      const int ds0 = (tid >> 6) * 32;
      const bf16_t* src = Vg + ((size_t)(j0 + jl) * BATCH + b) * DMODEL + col0 + ds0;
      v8bf x0 = *(const v8bf*)(src);
      v8bf x1 = *(const v8bf*)(src + 8);
      v8bf x2 = *(const v8bf*)(src + 16);
      v8bf x3 = *(const v8bf*)(src + 24);
#pragma unroll
      for (int q = 0; q < 8; ++q) {
        Vt[ds0 + q][jl]      = x0[q];
        Vt[ds0 + 8 + q][jl]  = x1[q];
        Vt[ds0 + 16 + q][jl] = x2[q];
        Vt[ds0 + 24 + q][jl] = x3[q];
      }
    }
    async_wait_all();
    __syncthreads();

    // ---- Sᵀ[64j][16i] : A = K rows (m=j), B = Q rows (n=i), K-dim = dk=64 --
    v8f st[4];
    {
      v8f z = {};
#pragma unroll
      for (int jt = 0; jt < 4; ++jt) st[jt] = z;
    }
#pragma unroll
    for (int jt = 0; jt < 4; ++jt) {
#pragma unroll
      for (int kt = 0; kt < 2; ++kt) {
        const v16bf a  = frag_a(&Ks[jt * 16][kt * 32], 64, lane);
        const v16bf bq = frag_b(&Qs[wave * 16][kt * 32], 64, lane);
        st[jt] = wmma_bf16f32(a, bq, st[jt]);
      }
    }

    // ---- softmax stats over j (M dim): lane holds 32 of 64 j's for its i --
    float sv[32];
    float mloc = -1e30f;
#pragma unroll
    for (int jt = 0; jt < 4; ++jt)
#pragma unroll
      for (int r = 0; r < 8; ++r) {
        const float x = st[jt][r] * 0.125f;   // 1/sqrt(dk=64)
        sv[jt * 8 + r] = x;
        mloc = fmaxf(mloc, x);
      }
    mloc = fmaxf(mloc, __shfl_xor(mloc, 16));   // combine half-waves (same i)
    const float mnew  = fmaxf(m_run, mloc);
    const float alpha = __expf(m_run - mnew);
    float lsum = 0.0f;
#pragma unroll
    for (int q = 0; q < 32; ++q) { sv[q] = __expf(sv[q] - mnew); lsum += sv[q]; }
    lsum += __shfl_xor(lsum, 16);
    l_run = l_run * alpha + lsum;
    m_run = mnew;

    // ---- write P (bf16) to per-wave LDS as [i][j] ----
    const int il   = lane & 15;
    const int jofs = (lane & 16) ? 8 : 0;
#pragma unroll
    for (int jt = 0; jt < 4; ++jt)
#pragma unroll
      for (int r = 0; r < 8; ++r)
        Ps[wave][il][jt * 16 + jofs + r] = (bf16_t)sv[jt * 8 + r];

    // ---- rescale running output ----
#pragma unroll
    for (int dt = 0; dt < 4; ++dt)
#pragma unroll
      for (int r = 0; r < 8; ++r) oacc[dt][r] *= alpha;

    // ---- Oᵀ[64d][16i] += Vᵀ · Pᵀ : A from Vt rows d, B from Ps rows i ----
#pragma unroll
    for (int dt = 0; dt < 4; ++dt) {
#pragma unroll
      for (int kt = 0; kt < 2; ++kt) {
        const v16bf a  = frag_a(&Vt[dt * 16][kt * 32], 64, lane);
        const v16bf bp = frag_b(&Ps[wave][0][kt * 32], 64, lane);
        oacc[dt] = wmma_bf16f32(a, bp, oacc[dt]);
      }
    }
  }

  // ---- epilogue: normalize and store context (bf16) ----
  const float inv_l = 1.0f / l_run;
  const int i     = i0 + wave * 16 + (lane & 15);
  const int dbase = (lane & 16) ? 8 : 0;
#pragma unroll
  for (int dt = 0; dt < 4; ++dt)
#pragma unroll
    for (int r = 0; r < 8; ++r) {
      const int d = col0 + dt * 16 + dbase + r;
      Og[((size_t)i * BATCH + b) * DMODEL + d] = (bf16_t)(oacc[dt][r] * inv_l);
    }
}

// ---------------------------------------------------------------------------
extern "C" void kernel_launch(void* const* d_in, const int* in_sizes, int n_in,
                              void* d_out, int out_size, void* d_ws, size_t ws_size,
                              hipStream_t stream) {
  (void)in_sizes; (void)n_in; (void)out_size; (void)ws_size;
  const float* query = (const float*)d_in[0];
  const float* key   = (const float*)d_in[1];
  const float* value = (const float*)d_in[2];
  const float* Wq    = (const float*)d_in[3];
  const float* bq    = (const float*)d_in[4];
  const float* Wk    = (const float*)d_in[5];
  const float* bk    = (const float*)d_in[6];
  const float* Wv    = (const float*)d_in[7];
  const float* bv    = (const float*)d_in[8];
  const float* Wo    = (const float*)d_in[9];
  const float* bo    = (const float*)d_in[10];
  float* out = (float*)d_out;

  // workspace: q, k, v, context — each MROWS*DMODEL bf16 (8 MB) = 32 MB total
  bf16_t* qb = (bf16_t*)d_ws;
  bf16_t* kb = qb + (size_t)MROWS * DMODEL;
  bf16_t* vb = kb + (size_t)MROWS * DMODEL;
  bf16_t* pb = vb + (size_t)MROWS * DMODEL;

  const dim3 blk(128);
  const dim3 gg(MROWS / 128, DMODEL / 64);      // 32 x 16 tiles

  gemm_xwt<false, false><<<gg, blk, 0, stream>>>(query, Wq, bq, qb, DMODEL, DMODEL);
  gemm_xwt<false, false><<<gg, blk, 0, stream>>>(key,   Wk, bk, kb, DMODEL, DMODEL);
  gemm_xwt<false, false><<<gg, blk, 0, stream>>>(value, Wv, bv, vb, DMODEL, DMODEL);

  const dim3 ga(S_LEN / 64, BATCH * NHEADS);    // 32 query tiles x 32 (b,h)
  flash_attn<<<ga, blk, 0, stream>>>(qb, kb, vb, pb);

  gemm_xwt<true, true><<<gg, blk, 0, stream>>>(pb, Wo, bo, out, DMODEL, DMODEL);
}
